// LinearAttention_87686052315975
// MI455X (gfx1250) — matrix operand order repaired and checked by reading, not compiled, not run on hardware
//
#include <hip/hip_runtime.h>

typedef unsigned short u16_t;
typedef __attribute__((ext_vector_type(8)))  unsigned short u16x8;
typedef __attribute__((ext_vector_type(16))) __bf16 v16bf;
typedef __attribute__((ext_vector_type(8)))  __bf16 v8bf;
typedef __attribute__((ext_vector_type(8)))  float  v8f;

// ---- helpers ---------------------------------------------------------------

__device__ __forceinline__ u16_t f2bf(float f) {
  union { float f; unsigned u; } c; c.f = f;
  return (u16_t)((c.u + 0x7FFFu + ((c.u >> 16) & 1u)) >> 16);  // RNE
}
__device__ __forceinline__ float bf2f(u16_t h) {
  union { unsigned u; float f; } c; c.u = ((unsigned)h) << 16;
  return c.f;
}
__device__ __forceinline__ void cp8(u16_t* dst, const u16_t* src) {
  *(u16x8*)dst = *(const u16x8*)src;
}

// A fragment (16x32 bf16, MxK): lanes 0-15 K 0-7/16-23, lanes 16-31 K 8-15/24-31
__device__ __forceinline__ v16bf load_a_frag(const u16_t* rowA, int lh) {
  const int kb = lh * 8;
  v8bf lo = *(const v8bf*)(rowA + kb);
  v8bf hi = *(const v8bf*)(rowA + 16 + kb);
  return __builtin_shufflevector(lo, hi, 0,1,2,3,4,5,6,7,8,9,10,11,12,13,14,15);
}
// B fragment (32x16 bf16, KxN): lane&15 = column, lanes 0-15 K 0-7/16-23, lanes 16-31 K 8-15/24-31
__device__ __forceinline__ v16bf load_b_frag(const u16_t* rowB, int lh) {
  const u16_t* p = rowB + lh * 8;
  v8bf lo = *(const v8bf*)(p);
  v8bf hi = *(const v8bf*)(p + 16);
  return __builtin_shufflevector(lo, hi, 0,1,2,3,4,5,6,7,8,9,10,11,12,13,14,15);
}
__device__ __forceinline__ v8f wmma_bf16(v16bf a, v16bf b, v8f c) {
  return __builtin_amdgcn_wmma_f32_16x16x32_bf16(false, a, false, b, (short)0, c,
                                                 false, false);
}

// ---- pass 0a: tiled transpose+convert  src[R][C] f32 -> dst[C][R] bf16 -----

__global__ __launch_bounds__(256) void tconv_f32(const float* __restrict__ src,
                                                 u16_t* __restrict__ dst,
                                                 int R, int C) {
  __shared__ float t[32][33];
  const size_t base = (size_t)blockIdx.z * R * C;
  const int r0 = blockIdx.y * 32, c0 = blockIdx.x * 32;
  const int tid = threadIdx.x;
  #pragma unroll
  for (int i = 0; i < 4; ++i) {
    int e = tid + i * 256, rr = e >> 5, cc = e & 31;
    t[rr][cc] = src[base + (size_t)(r0 + rr) * C + c0 + cc];
  }
  __syncthreads();
  #pragma unroll
  for (int i = 0; i < 4; ++i) {
    int e = tid + i * 256, cc = e >> 5, rr = e & 31;
    dst[base + (size_t)(c0 + cc) * R + r0 + rr] = f2bf(t[rr][cc]);
  }
}

// ---- pass 0b: plain convert f32 -> bf16 ------------------------------------

__global__ __launch_bounds__(256) void conv_bf16(const float* __restrict__ s,
                                                 u16_t* __restrict__ d, int n) {
  int i = blockIdx.x * 256 + threadIdx.x;
  if (i < n) d[i] = f2bf(s[i]);
}

// ---- pass 3b: tiled bf16 transpose  src[R][C] -> dst[C][R] -----------------

__global__ __launch_bounds__(256) void tpose_bf16(const u16_t* __restrict__ src,
                                                  u16_t* __restrict__ dst,
                                                  int R, int C) {
  __shared__ u16_t t[32][33];
  const size_t base = (size_t)blockIdx.z * R * C;
  const int r0 = blockIdx.y * 32, c0 = blockIdx.x * 32;
  const int tid = threadIdx.x;
  #pragma unroll
  for (int i = 0; i < 4; ++i) {
    int e = tid + i * 256, rr = e >> 5, cc = e & 31;
    t[rr][cc] = src[base + (size_t)(r0 + rr) * C + c0 + cc];
  }
  __syncthreads();
  #pragma unroll
  for (int i = 0; i < 4; ++i) {
    int e = tid + i * 256, cc = e >> 5, rr = e & 31;
    dst[base + (size_t)(c0 + cc) * R + r0 + rr] = t[rr][cc];
  }
}

// ---- kernel 1: qkv = w_qkv @ x   (M=1536, K=512, N=4096 per batch) ---------
// BM=BN=128, BK=32; 8 waves, each 64x32 (8 WMMAs / K-step).
// rows [0,512)->q bf16, [512,1024)->k bf16 (pre-softmax), [1024,1536)->v bf16

__global__ __launch_bounds__(256) void qkv_gemm(const u16_t* __restrict__ xT,
                                                const u16_t* __restrict__ wqkvb,
                                                u16_t* __restrict__ qb,
                                                u16_t* __restrict__ kb,
                                                u16_t* __restrict__ vb) {
  __shared__ __align__(16) u16_t Ash[128][40];
  __shared__ __align__(16) u16_t Bsh[128][40];
  const int tid = threadIdx.x;
  const int b  = blockIdx.z;
  const int m0 = blockIdx.y * 128;
  const int n0 = blockIdx.x * 128;
  const u16_t* xTb = xT + (size_t)b * 4096 * 512;   // [l][c]
  const int lane = tid & 31, wid = tid >> 5;
  const int wm = wid >> 2, wn = wid & 3;            // wave: 64 rows x 32 cols
  const int lm = lane & 15, lh = lane >> 4;
  const int sr = tid >> 1, skb = (tid & 1) * 16;    // staging: row, k-half
  v8f acc[4][2] = {};

  for (int k0 = 0; k0 < 512; k0 += 32) {
    const u16_t* asrc = wqkvb + (size_t)(m0 + sr) * 512 + k0 + skb;
    const u16_t* bsrc = xTb   + (size_t)(n0 + sr) * 512 + k0 + skb;
    cp8(&Ash[sr][skb],     asrc);
    cp8(&Ash[sr][skb + 8], asrc + 8);
    cp8(&Bsh[sr][skb],     bsrc);
    cp8(&Bsh[sr][skb + 8], bsrc + 8);
    __builtin_prefetch(asrc + 32, 0, 3);
    __builtin_prefetch(bsrc + 32, 0, 3);
    __syncthreads();
    v16bf af[4], bfr[2];
    #pragma unroll
    for (int tm = 0; tm < 4; ++tm)
      af[tm] = load_a_frag(&Ash[wm * 64 + tm * 16 + lm][0], lh);
    #pragma unroll
    for (int tn = 0; tn < 2; ++tn)
      bfr[tn] = load_b_frag(&Bsh[wn * 32 + tn * 16 + lm][0], lh);
    #pragma unroll
    for (int tm = 0; tm < 4; ++tm)
      #pragma unroll
      for (int tn = 0; tn < 2; ++tn)
        acc[tm][tn] = wmma_bf16(af[tm], bfr[tn], acc[tm][tn]);
    __syncthreads();
  }

  u16_t* qbb = qb + (size_t)b * 512 * 4096;
  u16_t* kbb = kb + (size_t)b * 512 * 4096;
  u16_t* vbb = vb + (size_t)b * 512 * 4096;
  #pragma unroll
  for (int tm = 0; tm < 4; ++tm)
    #pragma unroll
    for (int tn = 0; tn < 2; ++tn) {
      int mg = m0 + wm * 64 + tm * 16 + lh * 8;
      int ng = n0 + wn * 32 + tn * 16 + lm;
      #pragma unroll
      for (int r = 0; r < 8; ++r) {
        int o = mg + r;
        u16_t hv = f2bf(acc[tm][tn][r]);
        if (o < 512)        qbb[(size_t)o * 4096 + ng] = hv;
        else if (o < 1024)  kbb[(size_t)(o - 512) * 4096 + ng] = hv;
        else                vbb[(size_t)(o - 1024) * 4096 + ng] = hv;
      }
    }
}

// ---- kernel 2: softmax over l (4096) per k row, bf16 in place --------------

__global__ __launch_bounds__(256) void softmax_rows(u16_t* __restrict__ kbuf) {
  u16_t* p = kbuf + (size_t)blockIdx.x * 4096;
  const int tid = threadIdx.x;
  __shared__ float red[256];
  float mx = -3.4e38f;
  for (int i = tid; i < 4096; i += 256) mx = fmaxf(mx, bf2f(p[i]));
  red[tid] = mx; __syncthreads();
  for (int s = 128; s > 0; s >>= 1) {
    if (tid < s) red[tid] = fmaxf(red[tid], red[tid + s]);
    __syncthreads();
  }
  mx = red[0]; __syncthreads();
  float sum = 0.f;
  for (int i = tid; i < 4096; i += 256) sum += __expf(bf2f(p[i]) - mx);
  red[tid] = sum; __syncthreads();
  for (int s = 128; s > 0; s >>= 1) {
    if (tid < s) red[tid] += red[tid + s];
    __syncthreads();
  }
  const float inv = 1.0f / red[0];
  for (int i = tid; i < 4096; i += 256) p[i] = f2bf(__expf(bf2f(p[i]) - mx) * inv);
}

// ---- kernel 3: C[d,e] = sum_l k_sm[d,l] * v[e,l]  (64x64, K=4096) ----------
// output row-major [d][e] bf16 == B^T layout for kernel 4

__global__ __launch_bounds__(256) void context_gemm(const u16_t* __restrict__ kb,
                                                    const u16_t* __restrict__ vb,
                                                    u16_t* __restrict__ ctx) {
  __shared__ __align__(16) u16_t Ksh[64][40];
  __shared__ __align__(16) u16_t Vsh[64][40];
  const int tid = threadIdx.x;
  const int h = blockIdx.x, b = blockIdx.y;
  const u16_t* kp = kb + ((size_t)b * 512 + h * 64) * 4096;
  const u16_t* vp = vb + ((size_t)b * 512 + h * 64) * 4096;
  const int lane = tid & 31, wid = tid >> 5;
  const int wm = wid >> 2, wn = wid & 3;
  const int lm = lane & 15, lh = lane >> 4;
  const int sr = tid >> 2, skb = (tid & 3) * 8;
  v8f acc[2] = {};

  for (int l0 = 0; l0 < 4096; l0 += 32) {
    cp8(&Ksh[sr][skb], kp + (size_t)sr * 4096 + l0 + skb);
    cp8(&Vsh[sr][skb], vp + (size_t)sr * 4096 + l0 + skb);
    __syncthreads();
    v16bf bfr = load_b_frag(&Vsh[wn * 16 + lm][0], lh);
    acc[0] = wmma_bf16(load_a_frag(&Ksh[wm * 32 + lm][0], lh), bfr, acc[0]);
    acc[1] = wmma_bf16(load_a_frag(&Ksh[wm * 32 + 16 + lm][0], lh), bfr, acc[1]);
    __syncthreads();
  }

  u16_t* cp = ctx + ((size_t)b * 8 + h) * 64 * 64;
  #pragma unroll
  for (int tm = 0; tm < 2; ++tm) {
    int dg = wm * 32 + tm * 16 + lh * 8;
    int eg = wn * 16 + lm;
    #pragma unroll
    for (int r = 0; r < 8; ++r) cp[(dg + r) * 64 + eg] = f2bf(acc[tm][r]);
  }
}

// ---- kernel 4: W_eff[b][o][h*64+d] = sum_e w_out[o][h*64+e] * C[d][e] ------

__global__ __launch_bounds__(256) void weff_gemm(const u16_t* __restrict__ woutb,
                                                 const u16_t* __restrict__ ctx,
                                                 u16_t* __restrict__ weff) {
  __shared__ __align__(16) u16_t Ash[64][40];
  __shared__ __align__(16) u16_t Bsh[64][40];
  const int tid = threadIdx.x;
  const int m0 = blockIdx.x * 64;
  const int h = blockIdx.y, b = blockIdx.z;
  const u16_t* cp = ctx + ((size_t)b * 8 + h) * 64 * 64;
  const int lane = tid & 31, wid = tid >> 5;
  const int wm = wid >> 2, wn = wid & 3;
  const int lm = lane & 15, lh = lane >> 4;
  const int sr = tid >> 2, skb = (tid & 3) * 8;
  v8f acc[2] = {};

  for (int k0 = 0; k0 < 64; k0 += 32) {
    cp8(&Ash[sr][skb], woutb + (size_t)(m0 + sr) * 512 + h * 64 + k0 + skb);
    cp8(&Bsh[sr][skb], cp + sr * 64 + k0 + skb);
    __syncthreads();
    v16bf bfr = load_b_frag(&Bsh[wn * 16 + lm][0], lh);
    acc[0] = wmma_bf16(load_a_frag(&Ash[wm * 32 + lm][0], lh), bfr, acc[0]);
    acc[1] = wmma_bf16(load_a_frag(&Ash[wm * 32 + 16 + lm][0], lh), bfr, acc[1]);
    __syncthreads();
  }

  u16_t* wp = weff + (size_t)b * 512 * 512;
  #pragma unroll
  for (int tm = 0; tm < 2; ++tm) {
    int og = m0 + wm * 32 + tm * 16 + lh * 8;
    int dg = wn * 16 + lm;
    #pragma unroll
    for (int r = 0; r < 8; ++r)
      wp[(size_t)(og + r) * 512 + h * 64 + dg] = f2bf(acc[tm][r]);
  }
}

// ---- kernel 5: y = W_eff[b] @ q[b] + b_out  (M=512, K=512, N=4096) ---------

__global__ __launch_bounds__(256) void out_gemm(const u16_t* __restrict__ weff,
                                                const u16_t* __restrict__ qT,
                                                const float* __restrict__ bout,
                                                float* __restrict__ y) {
  __shared__ __align__(16) u16_t Ash[128][40];
  __shared__ __align__(16) u16_t Bsh[128][40];
  const int tid = threadIdx.x;
  const int b  = blockIdx.z;
  const int m0 = blockIdx.y * 128;
  const int n0 = blockIdx.x * 128;
  const u16_t* wb  = weff + (size_t)b * 512 * 512;   // [o][c]
  const u16_t* qTb = qT + (size_t)b * 4096 * 512;    // [l][c]
  float* yb = y + (size_t)b * 512 * 4096;
  const int lane = tid & 31, wid = tid >> 5;
  const int wm = wid >> 2, wn = wid & 3;
  const int lm = lane & 15, lh = lane >> 4;
  const int sr = tid >> 1, skb = (tid & 1) * 16;
  v8f acc[4][2] = {};

  for (int k0 = 0; k0 < 512; k0 += 32) {
    const u16_t* asrc = wb  + (size_t)(m0 + sr) * 512 + k0 + skb;
    const u16_t* bsrc = qTb + (size_t)(n0 + sr) * 512 + k0 + skb;
    cp8(&Ash[sr][skb],     asrc);
    cp8(&Ash[sr][skb + 8], asrc + 8);
    cp8(&Bsh[sr][skb],     bsrc);
    cp8(&Bsh[sr][skb + 8], bsrc + 8);
    __builtin_prefetch(asrc + 32, 0, 3);
    __builtin_prefetch(bsrc + 32, 0, 3);
    __syncthreads();
    v16bf af[4], bfr[2];
    #pragma unroll
    for (int tm = 0; tm < 4; ++tm)
      af[tm] = load_a_frag(&Ash[wm * 64 + tm * 16 + lm][0], lh);
    #pragma unroll
    for (int tn = 0; tn < 2; ++tn)
      bfr[tn] = load_b_frag(&Bsh[wn * 32 + tn * 16 + lm][0], lh);
    #pragma unroll
    for (int tm = 0; tm < 4; ++tm)
      #pragma unroll
      for (int tn = 0; tn < 2; ++tn)
        acc[tm][tn] = wmma_bf16(af[tm], bfr[tn], acc[tm][tn]);
    __syncthreads();
  }

  #pragma unroll
  for (int tm = 0; tm < 4; ++tm)
    #pragma unroll
    for (int tn = 0; tn < 2; ++tn) {
      int mg = m0 + wm * 64 + tm * 16 + lh * 8;
      int ng = n0 + wn * 32 + tn * 16 + lm;
      #pragma unroll
      for (int r = 0; r < 8; ++r)
        yb[(size_t)(mg + r) * 4096 + ng] = acc[tm][tn][r] + bout[mg + r];
    }
}

// ---- launch ----------------------------------------------------------------

extern "C" void kernel_launch(void* const* d_in, const int* in_sizes, int n_in,
                              void* d_out, int out_size, void* d_ws, size_t ws_size,
                              hipStream_t stream) {
  (void)in_sizes; (void)n_in; (void)out_size; (void)ws_size;
  const float* x    = (const float*)d_in[0];
  const float* wqkv = (const float*)d_in[1];
  const float* wout = (const float*)d_in[2];
  const float* bout = (const float*)d_in[3];
  float* y = (float*)d_out;

  const size_t NE = (size_t)16 * 512 * 4096;   // 33,554,432
  u16_t* xT    = (u16_t*)d_ws;                 // x^T  bf16 [b][l][c]
  u16_t* qb    = xT + NE;                      // q    bf16 [b][c][l]
  u16_t* qT    = qb + NE;                      // q^T  bf16 [b][l][c]
  u16_t* kbuf  = qT + NE;                      // k    bf16 [b][c][l] (softmax in place)
  u16_t* vb    = kbuf + NE;                    // v    bf16 [b][c][l]
  u16_t* ctx   = vb + NE;                      // context  [b][h][d][e]
  u16_t* weff  = ctx + (size_t)16 * 8 * 64 * 64;
  u16_t* wqkvb = weff + (size_t)16 * 512 * 512;
  u16_t* woutb = wqkvb + (size_t)1536 * 512;

  tconv_f32   <<<dim3(128, 16, 16), 256, 0, stream>>>(x, xT, 512, 4096);
  conv_bf16   <<<dim3(1536 * 512 / 256), 256, 0, stream>>>(wqkv, wqkvb, 1536 * 512);
  conv_bf16   <<<dim3(512 * 512 / 256),  256, 0, stream>>>(wout, woutb, 512 * 512);
  qkv_gemm    <<<dim3(32, 12, 16), 256, 0, stream>>>(xT, wqkvb, qb, kbuf, vb);
  softmax_rows<<<dim3(8192),       256, 0, stream>>>(kbuf);
  tpose_bf16  <<<dim3(128, 16, 16), 256, 0, stream>>>(qb, qT, 512, 4096);
  context_gemm<<<dim3(8, 16),      256, 0, stream>>>(kbuf, vb, ctx);
  weff_gemm   <<<dim3(8, 8, 16),   256, 0, stream>>>(woutb, ctx, weff);
  out_gemm    <<<dim3(32, 4, 16),  256, 0, stream>>>(weff, qT, bout, y);
}
